// CBOW_40836549050395
// MI455X (gfx1250) — matrix operand order repaired and checked by reading, not compiled, not run on hardware
//
#include <hip/hip_runtime.h>
#include <hip/hip_bf16.h>

#define VOC 50000
#define EMB 300
#define BATCH 256
#define CTX 6
#define NEGK 10

#define VO_ROWS   (BATCH)             // 256
#define VI_ROWS   (BATCH * CTX)       // 1536
#define NEG_ROWS  (BATCH * NEGK)      // 2560
#define TOT_ROWS  (VO_ROWS + VI_ROWS + NEG_ROWS)  // 4352

typedef __attribute__((ext_vector_type(16))) _Float16 v16h;
typedef __attribute__((ext_vector_type(8)))  float    v8f;

// ---------------------------------------------------------------------------
// Kernel 1: find the hot index of each one-hot row. ~870 MB of streaming
// traffic dominates the whole problem; float4 loads (global_load_b128),
// one block per row, early-exit once any lane in the block finds the 1.0.
// Padded vi rows are all-zero -> index stays -1 (reproduces the mask).
// ---------------------------------------------------------------------------
__global__ __launch_bounds__(256) void cbow_scan_kernel(
    const float* __restrict__ vo, const float* __restrict__ vi,
    const float* __restrict__ neg, int* __restrict__ idxbuf)
{
    __shared__ int s_idx;
    const int r = blockIdx.x;
    const float* row;
    if (r < VO_ROWS)                 row = vo  + (size_t)r * VOC;
    else if (r < VO_ROWS + VI_ROWS)  row = vi  + (size_t)(r - VO_ROWS) * VOC;
    else                             row = neg + (size_t)(r - VO_ROWS - VI_ROWS) * VOC;

    if (threadIdx.x == 0) s_idx = -1;
    __syncthreads();

    const float4* __restrict__ row4 = (const float4*)row;   // VOC = 12500 * 4
    volatile int* vflag = &s_idx;
    int found = -1;
    int it = 0;
    for (int i = threadIdx.x; i < VOC / 4; i += 256) {
        float4 v = row4[i];
        if (v.x != 0.0f)      found = 4 * i + 0;
        else if (v.y != 0.0f) found = 4 * i + 1;
        else if (v.z != 0.0f) found = 4 * i + 2;
        else if (v.w != 0.0f) found = 4 * i + 3;
        if (found >= 0) break;
        if (((++it) & 7) == 0 && *vflag >= 0) break;  // early exit, ~2x less HBM traffic
    }
    if (found >= 0) atomicExch(&s_idx, found);  // at most one lane ever writes
    __syncthreads();
    if (threadIdx.x == 0) idxbuf[r] = s_idx;
}

__device__ __forceinline__ float log_sigmoid(float x) {
    return fminf(x, 0.0f) - log1pf(__expf(-fabsf(x)));
}

// ---------------------------------------------------------------------------
// Kernel 2: one wave (32 lanes) per batch element. Build a 16x320 f16 tile:
//   row 0      = sum of valid context U-rows
//   rows 1..10 = U-rows of the 10 negatives
//   rows 11..15, cols 300..319 = zero padding
// Then score = tile @ vo_embed via v_wmma_f32_16x16x32_f16, with the B
// operand broadcast across N so column 0 of D holds all 11 dot products.
// ---------------------------------------------------------------------------
#define KPAD 320
#define MSTR 328   // row stride in halves (avoid systematic bank conflicts)

__global__ __launch_bounds__(32) void cbow_batch_kernel(
    const float* __restrict__ V, const float* __restrict__ U,
    const int* __restrict__ idx_vo, const int* __restrict__ idx_vi,
    const int* __restrict__ idx_neg, float* __restrict__ partial)
{
    __shared__ _Float16 sM[16][MSTR];
    __shared__ _Float16 sVo[KPAD + 16];
    __shared__ float    sScore[16];

    const int b = blockIdx.x;
    const int t = threadIdx.x;   // lane, wave32

    // zero the tile (covers padding rows/cols)
    for (int i = t; i < 16 * MSTR; i += 32) (&sM[0][0])[i] = (_Float16)0.0f;
    for (int i = t; i < KPAD + 16; i += 32) sVo[i] = (_Float16)0.0f;
    __syncthreads();

    // row 0: accumulate valid context embeddings (uniform branches per block)
    float cnt = 0.0f;
    float acc[10];
#pragma unroll
    for (int j = 0; j < 10; ++j) acc[j] = 0.0f;
    for (int c = 0; c < CTX; ++c) {
        const int ic = idx_vi[b * CTX + c];
        if (ic >= 0) {
            cnt += 1.0f;
            const float* __restrict__ Ur = U + (size_t)ic * EMB;
#pragma unroll
            for (int j = 0; j < 10; ++j) {
                const int d = t + 32 * j;
                if (d < EMB) acc[j] += Ur[d];
            }
        }
    }
#pragma unroll
    for (int j = 0; j < 10; ++j) {
        const int d = t + 32 * j;
        if (d < EMB) sM[0][d] = (_Float16)acc[j];
    }

    // rows 1..10: negative-sample embeddings
    for (int k = 0; k < NEGK; ++k) {
        const int ik = idx_neg[b * NEGK + k];
        const float* __restrict__ Ur = U + (size_t)ik * EMB;
#pragma unroll
        for (int j = 0; j < 10; ++j) {
            const int d = t + 32 * j;
            if (d < EMB) sM[1 + k][d] = (_Float16)Ur[d];
        }
    }

    // vo embedding
    {
        const int iv = idx_vo[b];
        const float* __restrict__ Vr = V + (size_t)iv * EMB;
#pragma unroll
        for (int j = 0; j < 10; ++j) {
            const int d = t + 32 * j;
            if (d < EMB) sVo[d] = (_Float16)Vr[d];
        }
    }
    __syncthreads();

    // WMMA: D(16x16 f32) += A(16x32 f16) * B(32x16 f16), K swept 0..319.
    // A layout (ISA 7.12.2, 16-bit A 16x32): lane m=t&15; lanes>=16 get K+8.
    // B broadcast across N: same K order as A (table order).
    const int m  = t & 15;
    const int hi = (t >> 4) << 3;   // 0 or 8
    v8f accm = {};
    for (int kb = 0; kb < KPAD; kb += 32) {
        v16h a, bb;
#pragma unroll
        for (int j = 0; j < 16; ++j) {
            const int ka = ((j >> 3) << 4) + hi + (j & 7);
            a[j] = sM[m][kb + ka];
        }
#pragma unroll
        for (int j = 0; j < 16; ++j) {
            const int kbx = ((j >> 3) << 4) + hi + (j & 7);   // same K order as A
            bb[j] = sVo[kb + kbx];
        }
        accm = __builtin_amdgcn_wmma_f32_16x16x32_f16(
            /*neg_a=*/false, a, /*neg_b=*/false, bb,
            /*c_mod=*/(short)0, accm, /*reuse_a=*/false, /*reuse_b=*/false);
    }

    // D layout: lanes 0-15 -> N=lane, M=vgpr; lanes 16-31 -> N=lane-16, M=vgpr+8.
    // Column N=0 (lanes 0 and 16) carries the 16 scores.
    if ((t & 15) == 0) {
#pragma unroll
        for (int g = 0; g < 8; ++g) sScore[hi + g] = accm[g];
    }
    __syncthreads();

    if (t == 0) {
        const float msum  = fmaxf(cnt, 1.0f);
        const float left  = log_sigmoid(sScore[0] / msum);
        float right = 0.0f;
#pragma unroll
        for (int k = 1; k <= NEGK; ++k) right += log_sigmoid(-sScore[k]);
        partial[b] = -(left + right);
    }
}

// ---------------------------------------------------------------------------
// Kernel 3: reduce 256 partial losses -> mean
// ---------------------------------------------------------------------------
__global__ __launch_bounds__(256) void cbow_reduce_kernel(
    const float* __restrict__ partial, float* __restrict__ out)
{
    __shared__ float s[256];
    const int t = threadIdx.x;
    s[t] = partial[t];
    __syncthreads();
    for (int off = 128; off > 0; off >>= 1) {
        if (t < off) s[t] += s[t + off];
        __syncthreads();
    }
    if (t == 0) out[0] = s[0] / (float)BATCH;
}

extern "C" void kernel_launch(void* const* d_in, const int* in_sizes, int n_in,
                              void* d_out, int out_size, void* d_ws, size_t ws_size,
                              hipStream_t stream) {
    const float* vo  = (const float*)d_in[0];  // [B, VOC]
    const float* vi  = (const float*)d_in[1];  // [B, CTX, VOC]
    const float* neg = (const float*)d_in[2];  // [B, K, VOC]
    const float* V   = (const float*)d_in[3];  // [VOC, EMB]
    const float* U   = (const float*)d_in[4];  // [VOC, EMB]

    int*   idxbuf  = (int*)d_ws;                                     // 4352 ints
    float* partial = (float*)((char*)d_ws + TOT_ROWS * sizeof(int)); // 256 floats

    cbow_scan_kernel<<<TOT_ROWS, 256, 0, stream>>>(vo, vi, neg, idxbuf);
    cbow_batch_kernel<<<BATCH, 32, 0, stream>>>(
        V, U, idxbuf, idxbuf + VO_ROWS, idxbuf + VO_ROWS + VI_ROWS, partial);
    cbow_reduce_kernel<<<1, 256, 0, stream>>>(partial, (float*)d_out);
}
